// GCN_78374563217803
// MI455X (gfx1250) — matrix operand order repaired and checked by reading, not compiled, not run on hardware
//
#include <hip/hip_runtime.h>
#include <hip/hip_bf16.h>
#include <math.h>

typedef __attribute__((ext_vector_type(16))) __bf16 v16bf;
typedef __attribute__((ext_vector_type(8)))  __bf16 v8bf;
typedef __attribute__((ext_vector_type(8)))  float  v8f;

// ---------------------------------------------------------------------------
// Tile MMA loop, unrolled at compile time so the reuse-A hint is a constant
// integer argument (frontend requirement for the WMMA builtin).
// ---------------------------------------------------------------------------
template<int NT, int LROW, int NF, int T = 0>
__device__ inline void mma_tiles(const v16bf& a, const __bf16* Wt,
                                 int k0, int hi, int m, v8f (&acc)[NT]) {
    const int col = T * 16 + m;
    const __bf16* wp = Wt + col * LROW + k0 + hi * 8;
    const v8bf blo = *(const v8bf*)(wp);
    const v8bf bhi = *(const v8bf*)(wp + 16);
    const v16bf b = __builtin_shufflevector(
        blo, bhi, 0, 1, 2, 3, 4, 5, 6, 7, 8, 9, 10, 11, 12, 13, 14, 15);
    acc[T] = __builtin_amdgcn_wmma_f32_16x16x32_bf16(
        false, a, false, b, (short)0, acc[T],
        /*reuse_a=*/(T + 1 < NT), /*reuse_b=*/false);
    if constexpr (T + 1 < NT)
        mma_tiles<NT, LROW, NF, T + 1>(a, Wt, k0, hi, m, acc);
}

// ---------------------------------------------------------------------------
// WMMA GEMM:  H[nrows, NF] = X[nrows, KD] @ W[KD, NF]      (bf16 in, f32 acc)
//
// Block = 256 threads = 8 waves; each wave owns 16 rows x (NT*16) cols.
// W is staged once per block into LDS, transposed to Wt[col][k] in bf16 with
// an 8-element row pad (row stride LROW):
//   - B fragment for lane (col = t*16 + m, hi = lane>>4) is two 8-bf16 runs
//     at Wt[col][k0 + hi*8 .. +8) and [k0 + 16 + hi*8 .. +8)  -> two 16B ds_load_b128, and the
//     8-bf16 pad shifts consecutive cols by 4 banks -> conflict-free.
// A fragment is two contiguous 8-float runs of the x row:
//   a[0..7]  <- x[row][k0 + hi*8      .. +8)   (two float4 loads)
//   a[8..15] <- x[row][k0 + 16 + hi*8 .. +8)   (two float4 loads)
// C/D (16x16 f32): VGPR r -> row = r + hi*8, col = lane&15.
// ---------------------------------------------------------------------------
template<int KD, int NF, int NT>
__launch_bounds__(256)
__global__ void gemm_wmma_bf16(const float* __restrict__ X,
                               const float* __restrict__ W,
                               float* __restrict__ H, int nrows) {
    constexpr int KDP  = (KD + 31) & ~31;   // K padded to WMMA depth
    constexpr int NFP  = NT * 16;           // cols padded to tile width
    constexpr int LROW = KDP + 8;           // padded LDS row stride (bf16)

    __shared__ __bf16 Wt[NFP * LROW];

    // ---- stage W -> LDS, transposed + bf16 (coalesced over columns) ----
    for (int idx = threadIdx.x; idx < NFP * KDP; idx += 256) {
        const int n = idx % NFP;
        const int k = idx / NFP;
        float w = 0.0f;
        if (n < NF && k < KD) w = W[k * NF + n];
        Wt[n * LROW + k] = (__bf16)w;
    }
    __syncthreads();

    const int lane = threadIdx.x & 31;
    const int wave = threadIdx.x >> 5;
    const int rowBase = (blockIdx.x * 8 + wave) * 16;
    const int m  = lane & 15;
    const int hi = lane >> 4;

    v8f acc[NT];
    #pragma unroll
    for (int t = 0; t < NT; ++t)
        #pragma unroll
        for (int j = 0; j < 8; ++j) acc[t][j] = 0.0f;

    int arow = rowBase + m;
    if (arow >= nrows) arow = nrows - 1;          // clamp loads; stores guarded
    const float* __restrict__ xrow = X + (long long)arow * KD;

    for (int k0 = 0; k0 < KDP; k0 += 32) {
        // ---- A fragment: 4 x global_load_b128 ----
        v16bf a;
        {
            const int kb0 = k0 + hi * 8;
            const int kb1 = k0 + 16 + hi * 8;
            float4 p0 = make_float4(0.f, 0.f, 0.f, 0.f), p1 = p0, p2 = p0, p3 = p0;
            if (kb0 + 8 <= KD) {
                p0 = *(const float4*)(xrow + kb0);
                p1 = *(const float4*)(xrow + kb0 + 4);
            }
            if (kb1 + 8 <= KD) {
                p2 = *(const float4*)(xrow + kb1);
                p3 = *(const float4*)(xrow + kb1 + 4);
            }
            a[0]  = (__bf16)p0.x;  a[1]  = (__bf16)p0.y;
            a[2]  = (__bf16)p0.z;  a[3]  = (__bf16)p0.w;
            a[4]  = (__bf16)p1.x;  a[5]  = (__bf16)p1.y;
            a[6]  = (__bf16)p1.z;  a[7]  = (__bf16)p1.w;
            a[8]  = (__bf16)p2.x;  a[9]  = (__bf16)p2.y;
            a[10] = (__bf16)p2.z;  a[11] = (__bf16)p2.w;
            a[12] = (__bf16)p3.x;  a[13] = (__bf16)p3.y;
            a[14] = (__bf16)p3.z;  a[15] = (__bf16)p3.w;
        }
        // ---- NT WMMAs sharing A (reuse-A hint on all but the last) ----
        mma_tiles<NT, LROW, NF>(a, &Wt[0], k0, hi, m, acc);
    }

    // ---- store D tiles ----
    #pragma unroll
    for (int t = 0; t < NT; ++t) {
        const int col = t * 16 + m;
        if (col >= NF) continue;
        #pragma unroll
        for (int r = 0; r < 8; ++r) {
            const int row = rowBase + hi * 8 + r;
            if (row < nrows) H[(long long)row * NF + col] = acc[t][r];
        }
    }
}

// ---------------------------------------------------------------------------
// Degree / norm kernels
// ---------------------------------------------------------------------------
__global__ void init_deg(float* __restrict__ deg, int n) {
    int i = blockIdx.x * blockDim.x + threadIdx.x;
    if (i < n) deg[i] = 1.0f;                     // self-loop
}

__global__ void count_deg(const long long* __restrict__ dst,
                          float* __restrict__ deg, int nE) {
    int e = blockIdx.x * blockDim.x + threadIdx.x;
    if (e < nE) atomicAdd(&deg[(int)dst[e]], 1.0f);
}

__global__ void finish_dinv(float* __restrict__ dinv, int n) {
    int i = blockIdx.x * blockDim.x + threadIdx.x;
    if (i < n) dinv[i] = rsqrtf(dinv[i]);
}

// ---------------------------------------------------------------------------
// Edge scatter: agg[dst] += h[src] * dinv[src]*dinv[dst]   (float4 per thread)
// ---------------------------------------------------------------------------
template<int F>
__global__ void scatter_edges(const long long* __restrict__ src,
                              const long long* __restrict__ dst,
                              const float* __restrict__ dinv,
                              const float* __restrict__ h,
                              float* __restrict__ agg, int nE) {
    constexpr int PER = F / 4;
    long long gid = (long long)blockIdx.x * blockDim.x + threadIdx.x;
    long long e  = gid / PER;
    int f4       = (int)(gid % PER);
    if (e >= nE) return;
    const int s = (int)src[e];
    const int d = (int)dst[e];
    const float norm = dinv[s] * dinv[d];
    const float4 hv = *(const float4*)(h + (long long)s * F + f4 * 4);
    float* ap = agg + (long long)d * F + f4 * 4;
    atomicAdd(ap + 0, hv.x * norm);
    atomicAdd(ap + 1, hv.y * norm);
    atomicAdd(ap + 2, hv.z * norm);
    atomicAdd(ap + 3, hv.w * norm);
}

// ---------------------------------------------------------------------------
// Finalize: agg += h * dinv^2 (self loop) + bias, optional ReLU (in place)
// ---------------------------------------------------------------------------
template<int F, bool RELU>
__global__ void finalize_layer(const float* __restrict__ h,
                               const float* __restrict__ dinv,
                               const float* __restrict__ bias,
                               float* __restrict__ agg, int n) {
    long long gid = (long long)blockIdx.x * blockDim.x + threadIdx.x;
    if (gid >= (long long)n * F) return;
    const int node = (int)(gid / F);
    const int f    = (int)(gid % F);
    const float di = dinv[node];
    float v = agg[gid] + h[gid] * di * di + bias[f];
    if (RELU) v = fmaxf(v, 0.0f);
    agg[gid] = v;
}

// ---------------------------------------------------------------------------
// log_softmax over 40 classes, one thread per node
// ---------------------------------------------------------------------------
__global__ void log_softmax40(const float* __restrict__ in,
                              float* __restrict__ out, int n) {
    int node = blockIdx.x * blockDim.x + threadIdx.x;
    if (node >= n) return;
    const float* v = in + (long long)node * 40;
    float mx = -INFINITY;
    #pragma unroll
    for (int j = 0; j < 40; ++j) mx = fmaxf(mx, v[j]);
    float s = 0.0f;
    #pragma unroll
    for (int j = 0; j < 40; ++j) s += __expf(v[j] - mx);
    const float ls = __logf(s) + mx;
    float* o = out + (long long)node * 40;
    #pragma unroll
    for (int j = 0; j < 40; ++j) o[j] = v[j] - ls;
}

// ---------------------------------------------------------------------------
// Launcher
// ---------------------------------------------------------------------------
extern "C" void kernel_launch(void* const* d_in, const int* in_sizes, int n_in,
                              void* d_out, int out_size, void* d_ws, size_t ws_size,
                              hipStream_t stream) {
    constexpr int IN = 512, H1 = 64, H2 = 16, OUT = 40;

    const float*     x   = (const float*)d_in[0];
    const long long* ei  = (const long long*)d_in[1];   // int64 edge_index [2, E]
    const float*     W1  = (const float*)d_in[2];
    const float*     b1  = (const float*)d_in[3];
    const float*     W2  = (const float*)d_in[4];
    const float*     b2  = (const float*)d_in[5];
    const float*     W3  = (const float*)d_in[6];
    const float*     b3  = (const float*)d_in[7];
    float*           out = (float*)d_out;

    const int N = in_sizes[0] / IN;
    const int E = in_sizes[1] / 2;
    const long long* src = ei;
    const long long* dst = ei + E;

    // workspace layout (f32)
    float* p    = (float*)d_ws;
    float* dinv = p;  p += N;
    float* h1   = p;  p += (size_t)N * H1;
    float* a1   = p;  p += (size_t)N * H1;
    float* h2   = p;  p += (size_t)N * H2;
    float* a2   = p;  p += (size_t)N * H2;
    float* h3   = p;  p += (size_t)N * OUT;
    float* a3   = p;  p += (size_t)N * OUT;

    const int B = 256;
    auto cdiv = [](long long a, long long b) { return (int)((a + b - 1) / b); };

    // ---- degrees / norm ----
    init_deg  <<<cdiv(N, B), B, 0, stream>>>(dinv, N);
    count_deg <<<cdiv(E, B), B, 0, stream>>>(dst, dinv, E);
    finish_dinv<<<cdiv(N, B), B, 0, stream>>>(dinv, N);

    const int gemm_blocks = cdiv(N, 128);

    // ---- layer 1: 512 -> 64, ReLU ----
    (void)hipMemsetAsync(a1, 0, (size_t)N * H1 * sizeof(float), stream);
    gemm_wmma_bf16<IN, H1, H1 / 16><<<gemm_blocks, 256, 0, stream>>>(x, W1, h1, N);
    scatter_edges<H1><<<cdiv((long long)E * (H1 / 4), B), B, 0, stream>>>(src, dst, dinv, h1, a1, E);
    finalize_layer<H1, true><<<cdiv((long long)N * H1, B), B, 0, stream>>>(h1, dinv, b1, a1, N);

    // ---- layer 2: 64 -> 16, ReLU ----
    (void)hipMemsetAsync(a2, 0, (size_t)N * H2 * sizeof(float), stream);
    gemm_wmma_bf16<H1, H2, 1><<<gemm_blocks, 256, 0, stream>>>(a1, W2, h2, N);
    scatter_edges<H2><<<cdiv((long long)E * (H2 / 4), B), B, 0, stream>>>(src, dst, dinv, h2, a2, E);
    finalize_layer<H2, true><<<cdiv((long long)N * H2, B), B, 0, stream>>>(h2, dinv, b2, a2, N);

    // ---- layer 3: 16 -> 40 (K padded to 32 inside the GEMM) ----
    (void)hipMemsetAsync(a3, 0, (size_t)N * OUT * sizeof(float), stream);
    gemm_wmma_bf16<H2, OUT, 3><<<gemm_blocks, 256, 0, stream>>>(a2, W3, h3, N);
    scatter_edges<OUT><<<cdiv((long long)E * (OUT / 4), B), B, 0, stream>>>(src, dst, dinv, h3, a3, E);
    finalize_layer<OUT, false><<<cdiv((long long)N * OUT, B), B, 0, stream>>>(h3, dinv, b3, a3, N);

    // ---- log_softmax ----
    log_softmax40<<<cdiv(N, B), B, 0, stream>>>(a3, out, N);
}
